// SelfAttention_51977694216840
// MI455X (gfx1250) — matrix operand order repaired and checked by reading, not compiled, not run on hardware
//
#include <hip/hip_runtime.h>

// Problem constants (from reference)
#define E_DIM 1024
#define S_LEN 2048
#define BATCH 2
#define NHEAD 16
#define HDIM  64
#define NTOK  (BATCH * S_LEN)   // 4096

typedef __attribute__((ext_vector_type(16))) __bf16 v16bf;
typedef __attribute__((ext_vector_type(8)))  float  v8f;
typedef __attribute__((ext_vector_type(4)))  int    v4i;
typedef __attribute__((address_space(3))) v4i* lds_v4i_t;

union Frag16 { unsigned u[8]; v16bf v; };

#if defined(__has_builtin)
#if __has_builtin(__builtin_amdgcn_global_load_async_to_lds_b128)
#define HAS_ASYNC_LDS 1
#endif
#if __has_builtin(__builtin_amdgcn_s_wait_asynccnt)
#define HAS_WAIT_ASYNC 1
#endif
#endif

// f32 -> bf16 bits, round-to-nearest-even
__device__ __forceinline__ unsigned short f2bf(float x) {
  unsigned u = __builtin_bit_cast(unsigned, x);
  u += 0x7FFFu + ((u >> 16) & 1u);
  return (unsigned short)(u >> 16);
}
__device__ __forceinline__ unsigned pack2bf(float lo, float hi) {
  return (unsigned)f2bf(lo) | ((unsigned)f2bf(hi) << 16);
}
// 16-bit A-matrix (16x32) pair index within a 32-wide K row, contiguous order:
// lane<16: VGPR v -> K pairs {0,2,...,14}; lane>=16: +16
__device__ __forceinline__ int pairIdxA(int v, int half) {
  return v + half * 8;
}

// -------------------------------------------------------------------------
// Kernel 1: C = x @ W + b, output quantized to bf16 in attention-friendly
// layouts.  vlayout==0 : dst[b][h][s][d]   (Q, K)
//           vlayout==1 : dst[b][h][d][s]   (V transposed, so PV B-fragments
//                                           have contiguous key pairs)
// Block: 256 threads (8 waves). Tile: 128(M) x 64(N), K-step 32.
// -------------------------------------------------------------------------
__global__ __launch_bounds__(256)
void qkv_gemm_bf16(const float* __restrict__ x, const float* __restrict__ W,
                   const float* __restrict__ bias,
                   unsigned short* __restrict__ dst, int vlayout) {
  __shared__ unsigned short As16[128 * 32];   // A tile, row-major bf16
  __shared__ unsigned short Bs16[64 * 32];    // B tile, stored [n][k]
  unsigned* AsU = (unsigned*)As16;
  const unsigned* AsUc = (const unsigned*)As16;
  const unsigned* BsUc = (const unsigned*)Bs16;

  const int t    = threadIdx.x;
  const int m0   = blockIdx.x * 128;
  const int n0   = blockIdx.y * 64;
  const int w    = t >> 5;
  const int lane = t & 31;
  const int half = lane >> 4;
  const int lr   = lane & 15;
  const int wm   = (w & 3) * 32;   // wave row offset in block tile
  const int wn   = (w >> 2) * 32;  // wave col offset in block tile

  v8f c[2][2] = {};

  for (int k0 = 0; k0 < E_DIM; k0 += 32) {
    // A: 128x32 f32 -> 2048 bf16-pairs, 8 per thread (pair load = float2)
    #pragma unroll
    for (int i = 0; i < 8; ++i) {
      int p = t + 256 * i;            // pair index 0..2047
      int r = p >> 4, cp = p & 15;
      const float* g = x + (long)(m0 + r) * E_DIM + k0 + cp * 2;
      AsU[r * 16 + cp] = pack2bf(g[0], g[1]);
    }
    // B: W[k][n] (n contiguous, coalesced) -> Bs[n][k]
    #pragma unroll
    for (int i = 0; i < 8; ++i) {
      int idx = t + 256 * i;          // 0..2047
      int kk = idx >> 6, nn = idx & 63;
      Bs16[nn * 32 + kk] = f2bf(W[(long)(k0 + kk) * E_DIM + n0 + nn]);
    }
    __syncthreads();

    Frag16 a[2], b[2];
    #pragma unroll
    for (int sm = 0; sm < 2; ++sm) {
      int row = wm + sm * 16 + lr;
      #pragma unroll
      for (int v = 0; v < 8; ++v) a[sm].u[v] = AsUc[row * 16 + pairIdxA(v, half)];
    }
    #pragma unroll
    for (int sn = 0; sn < 2; ++sn) {
      int col = wn + sn * 16 + lr;
      #pragma unroll
      for (int v = 0; v < 8; ++v) b[sn].u[v] = BsUc[col * 16 + v + half * 8];
    }
    #pragma unroll
    for (int sm = 0; sm < 2; ++sm)
      #pragma unroll
      for (int sn = 0; sn < 2; ++sn)
        c[sm][sn] = __builtin_amdgcn_wmma_f32_16x16x32_bf16(
            false, a[sm].v, false, b[sn].v, (short)0, c[sm][sn], false, false);
    __syncthreads();
  }

  // bias + store (C layout: VGPR j -> row j + 8*half; col = lr)
  #pragma unroll
  for (int sn = 0; sn < 2; ++sn) {
    int n_g = n0 + wn + sn * 16 + lr;
    float bv = bias[n_g];
    int h = n_g >> 6, d = n_g & 63;
    #pragma unroll
    for (int sm = 0; sm < 2; ++sm) {
      #pragma unroll
      for (int j = 0; j < 8; ++j) {
        int m_g = m0 + wm + sm * 16 + j + half * 8;
        int bb = m_g >> 11, s = m_g & (S_LEN - 1);
        float val = c[sm][sn][j] + bv;
        long addr;
        if (vlayout == 0)
          addr = (((long)(bb * NHEAD + h) * S_LEN + s) * HDIM + d);
        else
          addr = (((long)(bb * NHEAD + h) * HDIM + d) * S_LEN + s);
        dst[addr] = f2bf(val);
      }
    }
  }
}

// -------------------------------------------------------------------------
// Kernel 2: flash-style attention with workgroup-shared K/V tiles.
// Block: 128 threads = 4 waves, all on the same (b,h); wave wv handles the
// 16-query block q0 = qc*64 + wv*16.  Key loop: 64 keys / iteration, K/V
// tiles staged in LDS (async-to-LDS when the toolchain exposes it).
// Q,K: bf16 [B,H,S,D]; V: bf16 [B,H,D,S]; out: f32 [B,S,E].
// -------------------------------------------------------------------------
__global__ __launch_bounds__(128)
void attention_fwd(const unsigned short* __restrict__ Q,
                   const unsigned short* __restrict__ K,
                   const unsigned short* __restrict__ V,
                   float* __restrict__ out) {
  __shared__ uint4 Ks4[512];                 // 64 keys x 64 d   (8 KB)
  __shared__ uint4 Vs4[512];                 // 64 d x 64 keys   (8 KB, V^T)
  __shared__ unsigned short Pw[4][16 * 64];  // per-wave P tile  (8 KB)

  const unsigned* KsU = (const unsigned*)Ks4;
  const unsigned* VsU = (const unsigned*)Vs4;

  const int t    = threadIdx.x;
  const int wv   = t >> 5;
  const int lane = t & 31;
  const int half = lane >> 4;
  const int lr   = lane & 15;

  const int bh = blockIdx.x >> 5;  // (b*H + h), 0..31
  const int qc = blockIdx.x & 31;  // 64-query chunk, 0..31
  const int b  = bh >> 4, h = bh & 15;
  const int q0 = qc * 64 + wv * 16;

  const unsigned* Qbh = (const unsigned*)(Q + (long)bh * S_LEN * HDIM);
  const uint4* KbhG = (const uint4*)(K + (long)bh * S_LEN * HDIM);
  const uint4* VbhG = (const uint4*)(V + (long)bh * HDIM * S_LEN);

  // Q A-fragments: 16 queries x D=64 -> two 16x32 fragments (dd0 = 0, 32)
  Frag16 qa[2];
  {
    const unsigned* qr = Qbh + (q0 + lr) * (HDIM / 2);  // 32 dwords / row
    #pragma unroll
    for (int f = 0; f < 2; ++f)
      #pragma unroll
      for (int v = 0; v < 8; ++v)
        qa[f].u[v] = qr[f * 16 + pairIdxA(v, half)];
  }

  float m_i[8], l_i[8];
  #pragma unroll
  for (int j = 0; j < 8; ++j) { m_i[j] = -3.0e38f; l_i[j] = 0.0f; }
  v8f o[4] = {};

  const float LOG2E = 1.4426950408889634f;
  unsigned short* Pw_w = &Pw[wv][0];
  const unsigned* PwU  = (const unsigned*)&Pw[wv][0];

  for (int kb = 0; kb < S_LEN / 64; ++kb) {
    const int key0 = kb * 64;

    // ---- cooperative K/V tile staging: 2x 512 uint4, 4 each per thread ----
    #pragma unroll
    for (int i = 0; i < 4; ++i) {
      int idx = t + 128 * i;            // 0..511
      int row = idx >> 3, off = idx & 7;
      const uint4* gk = KbhG + (long)(key0 + row) * 8 + off;           // K[key][d]
      const uint4* gv = VbhG + (long)row * (S_LEN / 8) + (key0 >> 3) + off; // V^T[d][s]
#ifdef HAS_ASYNC_LDS
      __builtin_amdgcn_global_load_async_to_lds_b128(
          (v4i*)gk, (lds_v4i_t)(Ks4 + idx), 0, 0);
      __builtin_amdgcn_global_load_async_to_lds_b128(
          (v4i*)gv, (lds_v4i_t)(Vs4 + idx), 0, 0);
#else
      Ks4[idx] = *gk;
      Vs4[idx] = *gv;
#endif
    }
#ifdef HAS_ASYNC_LDS
#ifdef HAS_WAIT_ASYNC
    __builtin_amdgcn_s_wait_asynccnt(0);
#else
    asm volatile("s_wait_asynccnt 0x0" ::: "memory");
#endif
#endif
    __syncthreads();

    // ---- scores: 4 key-tiles of 16, each contracting D=64 in two chunks ----
    v8f st[4];
    #pragma unroll
    for (int ti = 0; ti < 4; ++ti) {
      Frag16 kf0, kf1;
      int krow = ti * 16 + lr;
      #pragma unroll
      for (int v = 0; v < 8; ++v) {
        kf0.u[v] = KsU[krow * 32 + half * 8 + v];       // dd 0..31
        kf1.u[v] = KsU[krow * 32 + 16 + half * 8 + v];  // dd 32..63
      }
      v8f z = {};
      z = __builtin_amdgcn_wmma_f32_16x16x32_bf16(false, qa[0].v, false, kf0.v, (short)0, z, false, false);
      z = __builtin_amdgcn_wmma_f32_16x16x32_bf16(false, qa[1].v, false, kf1.v, (short)0, z, false, false);
      st[ti] = z;
    }

    // ---- online softmax over 64 keys (exp2 domain), per row j ----
    #pragma unroll
    for (int j = 0; j < 8; ++j) {
      float s0 = st[0][j] * LOG2E, s1 = st[1][j] * LOG2E;
      float s2 = st[2][j] * LOG2E, s3 = st[3][j] * LOG2E;
      float mx = fmaxf(fmaxf(s0, s1), fmaxf(s2, s3));
      #pragma unroll
      for (int d = 1; d < 16; d <<= 1) mx = fmaxf(mx, __shfl_xor(mx, d, 32));
      float m_new = fmaxf(m_i[j], mx);
      float scale = exp2f(m_i[j] - m_new);
      float p0 = exp2f(s0 - m_new), p1 = exp2f(s1 - m_new);
      float p2 = exp2f(s2 - m_new), p3 = exp2f(s3 - m_new);
      float rs = (p0 + p1) + (p2 + p3);
      #pragma unroll
      for (int d = 1; d < 16; d <<= 1) rs += __shfl_xor(rs, d, 32);
      l_i[j] = l_i[j] * scale + rs;
      m_i[j] = m_new;
      #pragma unroll
      for (int dt = 0; dt < 4; ++dt) o[dt][j] *= scale;
      int row = j + half * 8;
      Pw_w[row * 64 + lr]      = f2bf(p0);
      Pw_w[row * 64 + 16 + lr] = f2bf(p1);
      Pw_w[row * 64 + 32 + lr] = f2bf(p2);
      Pw_w[row * 64 + 48 + lr] = f2bf(p3);
    }
    // wave-private LDS C->A reshape; LDS is in-order within a wave, block
    // compiler motion across the store->load boundary
    __builtin_amdgcn_sched_barrier(0);
    Frag16 pa[2];
    #pragma unroll
    for (int c2 = 0; c2 < 2; ++c2)
      #pragma unroll
      for (int v = 0; v < 8; ++v)
        pa[c2].u[v] = PwU[lr * 32 + c2 * 16 + pairIdxA(v, half)];
    __builtin_amdgcn_sched_barrier(0);

    // ---- PV: O += P(16x64) @ V(64x64), B-fragments from V^T tile ----
    #pragma unroll
    for (int dt = 0; dt < 4; ++dt) {
      Frag16 vf0, vf1;
      int vrow = dt * 16 + lr;
      #pragma unroll
      for (int v = 0; v < 8; ++v) {
        vf0.u[v] = VsU[vrow * 32 + half * 8 + v];       // keys 0..31
        vf1.u[v] = VsU[vrow * 32 + 16 + half * 8 + v];  // keys 32..63
      }
      o[dt] = __builtin_amdgcn_wmma_f32_16x16x32_bf16(false, pa[0].v, false, vf0.v, (short)0, o[dt], false, false);
      o[dt] = __builtin_amdgcn_wmma_f32_16x16x32_bf16(false, pa[1].v, false, vf1.v, (short)0, o[dt], false, false);
    }
    __syncthreads();  // protect Ks/Vs before next iteration's staging
  }

  // normalize and store f32 to [B, S, E]
  #pragma unroll
  for (int dt = 0; dt < 4; ++dt) {
    int colE = h * HDIM + dt * 16 + lr;
    #pragma unroll
    for (int j = 0; j < 8; ++j) {
      int s = q0 + j + half * 8;
      out[((long)b * S_LEN + s) * E_DIM + colE] = o[dt][j] / l_i[j];
    }
  }
}

// -------------------------------------------------------------------------
extern "C" void kernel_launch(void* const* d_in, const int* in_sizes, int n_in,
                              void* d_out, int out_size, void* d_ws, size_t ws_size,
                              hipStream_t stream) {
  const float* x  = (const float*)d_in[0];
  const float* Wq = (const float*)d_in[1];
  const float* bq = (const float*)d_in[2];
  const float* Wk = (const float*)d_in[3];
  const float* bk = (const float*)d_in[4];
  const float* Wv = (const float*)d_in[5];
  const float* bv = (const float*)d_in[6];
  float* out = (float*)d_out;

  // Workspace: Q,K,V bf16, 4M elements (8 MB) each -> 24 MB total
  const long QKV_ELEMS = (long)BATCH * NHEAD * S_LEN * HDIM;  // 4194304
  unsigned short* Qws = (unsigned short*)d_ws;
  unsigned short* Kws = Qws + QKV_ELEMS;
  unsigned short* Vws = Kws + QKV_ELEMS;

  dim3 g1(NTOK / 128, E_DIM / 64);  // (32, 16)
  qkv_gemm_bf16<<<g1, 256, 0, stream>>>(x, Wq, bq, Qws, 0);
  qkv_gemm_bf16<<<g1, 256, 0, stream>>>(x, Wk, bk, Kws, 0);
  qkv_gemm_bf16<<<g1, 256, 0, stream>>>(x, Wv, bv, Vws, 1);

  // 32 (b,h) x 32 query-chunks of 64 -> 1024 blocks x 4 waves
  attention_fwd<<<1024, 128, 0, stream>>>(Qws, Kws, Vws, out);
}
